// CausalSelfAttention_66967130079464
// MI455X (gfx1250) — matrix operand order repaired and checked by reading, not compiled, not run on hardware
//
#include <hip/hip_runtime.h>
#include <hip/hip_bf16.h>

// ---------------------------------------------------------------------------
// CDNA5 (gfx1250) causal self-attention, bf16 WMMA path, wave32.
// B=2, S=2048, H=1024, NH=16, HD=64.
// TDM (tensor_load_to_lds) stages K/W tiles; permlane16 softmax reductions.
// ---------------------------------------------------------------------------

typedef __attribute__((ext_vector_type(4)))  __bf16 v4bf;
typedef __attribute__((ext_vector_type(8)))  __bf16 v8bf;
typedef __attribute__((ext_vector_type(16))) __bf16 v16bf;
typedef __attribute__((ext_vector_type(8)))  float  v8f;
typedef __attribute__((ext_vector_type(4)))  unsigned int v4u;
typedef __attribute__((ext_vector_type(8)))  int v8i;
typedef __attribute__((ext_vector_type(4)))  int v4i;

union FragAB { v16bf v; v8bf h[2]; };

#define WMMA_BF16(Af, Bf, Cf) \
  __builtin_amdgcn_wmma_f32_16x16x32_bf16(false, (Af), false, (Bf), (short)0, (Cf), false, false)

// Problem constants
#define C_B   2
#define C_S   2048
#define C_H   1024
#define C_NH  16
#define C_HD  64
#define LOG_THETA 9.210340371976184f   // ln(10000)

// ---------------------------------------------------------------------------
// TDM: issue a 2D bf16 tile load (rows x cols) global -> LDS with optional
// LDS row padding. D# bitfields per CDNA5 ISA ch.8 (groups 0/1; 2D so groups
// 2/3 are zero). pad_interval/pad_amount are the encoded field values.
// This toolchain exposes the 6-arg builtin (extra int32x8 group before cpol).
// ---------------------------------------------------------------------------
__device__ __forceinline__
void tdm_load_tile_bf16(unsigned lds_off, const void* gptr,
                        unsigned tensor_d0, unsigned tensor_d1, unsigned stride0,
                        unsigned tile_d0, unsigned tile_d1,
                        unsigned pad_interval, unsigned pad_amount, bool pad_en) {
  const unsigned long long ga = (unsigned long long)gptr;
  v4u g0;
  g0[0] = 1u;                                             // count=1, user desc
  g0[1] = lds_off;                                        // LDS byte address
  g0[2] = (unsigned)ga;                                   // global_addr[31:0]
  g0[3] = ((unsigned)(ga >> 32) & 0x01FFFFFFu) | (2u << 30); // addr[56:32]|type=2
  unsigned d0 = (1u << 16);                               // data_size = 2 bytes
  if (pad_en) d0 |= (1u << 20) | (pad_interval << 22) | (pad_amount << 25);
  v8i g1;
  g1[0] = (int)d0;                                        // wg_mask=0 (no cluster)
  g1[1] = (int)((tensor_d0 & 0xFFFFu) << 16);             // bar_addr=0 | dim0.lo
  g1[2] = (int)(((tensor_d0 >> 16) & 0xFFFFu) | ((tensor_d1 & 0xFFFFu) << 16));
  g1[3] = (int)(((tensor_d1 >> 16) & 0xFFFFu) | (tile_d0 << 16));
  g1[4] = (int)(tile_d1 & 0xFFFFu);                       // tile_dim1 | tile_dim2=0
  g1[5] = (int)stride0;                                   // dim0 stride (elements)
  g1[6] = 0;
  g1[7] = 0;
  const v4i z4 = {0, 0, 0, 0};
  const v8i z8 = {0, 0, 0, 0, 0, 0, 0, 0};
  __builtin_amdgcn_tensor_load_to_lds(g0, g1, z4, z4, z8, 0);
}

// ---------------------------------------------------------------------------
// VALU butterfly reductions within 16-lane rows (v_permlane16_b32).
// sel nibble for lane i = i ^ m, packed low-lane-first.
// ---------------------------------------------------------------------------
__device__ __forceinline__ float permlane16_xor(float x, unsigned s0, unsigned s1) {
  const unsigned u = __builtin_bit_cast(unsigned, x);
  return __builtin_bit_cast(float, __builtin_amdgcn_permlane16(u, u, s0, s1, false, false));
}
__device__ __forceinline__ float row16_max(float x) {
  x = fmaxf(x, permlane16_xor(x, 0x67452301u, 0xEFCDAB89u));  // xor 1
  x = fmaxf(x, permlane16_xor(x, 0x45670123u, 0xCDEF89ABu));  // xor 3
  x = fmaxf(x, permlane16_xor(x, 0x01234567u, 0x89ABCDEFu));  // xor 7
  x = fmaxf(x, permlane16_xor(x, 0xFEDCBA98u, 0x76543210u));  // xor 8
  return x;
}
__device__ __forceinline__ float row16_sum(float x) {
  x += permlane16_xor(x, 0x67452301u, 0xEFCDAB89u);
  x += permlane16_xor(x, 0x45670123u, 0xCDEF89ABu);
  x += permlane16_xor(x, 0x01234567u, 0x89ABCDEFu);
  x += permlane16_xor(x, 0xFEDCBA98u, 0x76543210u);
  return x;
}

// ---------------------------------------------------------------------------
// fp32 -> bf16 conversion (vectorized, n4 = count/4)
// ---------------------------------------------------------------------------
__global__ __launch_bounds__(256)
void convert_f32_bf16(const float* __restrict__ src, __bf16* __restrict__ dst, int n4) {
  int i = blockIdx.x * blockDim.x + threadIdx.x;
  if (i < n4) {
    const float4 f = ((const float4*)src)[i];
    v4bf o;
    o.x = (__bf16)f.x; o.y = (__bf16)f.y; o.z = (__bf16)f.z; o.w = (__bf16)f.w;
    ((v4bf*)dst)[i] = o;
  }
}

// ---------------------------------------------------------------------------
// GEMM: C[M=4096, N=1024] = A[M,1024] @ W[N,1024]^T   (bf16 in, f32 acc)
// Block = 256 threads (8 waves), tile 128x64; wave -> 16 rows x 64 cols.
// W tile (64x64 per K-chunk) staged into LDS by the TDM, padded to 144 B rows.
// EPI 0: fp32 out [M,N];  EPI 1: RoPE + bf16 out [B,NH,S,HD];  EPI 2: bf16 out.
// ---------------------------------------------------------------------------
template<int EPI>
__global__ __launch_bounds__(256)
void gemm_bf16(const __bf16* __restrict__ A, const __bf16* __restrict__ W,
               float* __restrict__ Cf, __bf16* __restrict__ Cb) {
  constexpr int Kd = C_H;                 // 1024
  __shared__ __bf16 ldsW[64 * 72];        // W tile 64x64, padded stride 72

  const int tid    = threadIdx.x;
  const int lane   = tid & 31;
  const int wv     = tid >> 5;
  const int lane16 = lane & 15;
  const int hi8    = (lane >> 4) * 8;     // K-run offset (A) / row offset (C/D)
  const int krun   = (lane >> 4) * 16;    // unused (B fragments use hi8)

  const int m0 = blockIdx.x * 128 + wv * 16;
  const int n0 = blockIdx.y * 64;
  const unsigned ldsW_off = (unsigned)(size_t)(void*)ldsW;

  v8f acc0 = {}, acc1 = {}, acc2 = {}, acc3 = {};
  const __bf16* arow = A + (size_t)(m0 + lane16) * Kd;

  for (int k0 = 0; k0 < Kd; k0 += 64) {
    __syncthreads();
    if (wv == 0) {  // TDM DMA: 64 rows x 64 cols bf16, pad 128B rows by 16B
      tdm_load_tile_bf16(ldsW_off, W + (size_t)n0 * Kd + k0,
                         /*tensor_d0=*/Kd, /*tensor_d1=*/C_H, /*stride0=*/Kd,
                         /*tile_d0=*/64, /*tile_d1=*/64,
                         /*pad_interval(32dw)=*/4u, /*pad_amount(4dw)=*/3u, true);
      __builtin_amdgcn_s_wait_tensorcnt(0);
    }
    __syncthreads();

    if (k0 + 64 < Kd) __builtin_prefetch(arow + k0 + 64, 0, 0);

#pragma unroll
    for (int ks = 0; ks < 2; ++ks) {
      const int kk = k0 + ks * 32;
      FragAB a;                                  // A fragment: 16x32 bf16
      a.h[0] = *(const v8bf*)(arow + kk + hi8);
      a.h[1] = *(const v8bf*)(arow + kk + 16 + hi8);

      FragAB b0, b1, b2, b3;                     // B fragments: 32x16 each
      const int kb = ks * 32 + hi8;
      b0.h[0] = *(const v8bf*)&ldsW[( 0 + lane16) * 72 + kb];
      b0.h[1] = *(const v8bf*)&ldsW[( 0 + lane16) * 72 + kb + 16];
      b1.h[0] = *(const v8bf*)&ldsW[(16 + lane16) * 72 + kb];
      b1.h[1] = *(const v8bf*)&ldsW[(16 + lane16) * 72 + kb + 16];
      b2.h[0] = *(const v8bf*)&ldsW[(32 + lane16) * 72 + kb];
      b2.h[1] = *(const v8bf*)&ldsW[(32 + lane16) * 72 + kb + 16];
      b3.h[0] = *(const v8bf*)&ldsW[(48 + lane16) * 72 + kb];
      b3.h[1] = *(const v8bf*)&ldsW[(48 + lane16) * 72 + kb + 16];

      acc0 = WMMA_BF16(a.v, b0.v, acc0);
      acc1 = WMMA_BF16(a.v, b1.v, acc1);
      acc2 = WMMA_BF16(a.v, b2.v, acc2);
      acc3 = WMMA_BF16(a.v, b3.v, acc3);
    }
  }

  // Epilogue. D layout: VGPR j -> row m0+j+hi8, col = n0 + t*16 + lane16.
  if constexpr (EPI == 0) {
#pragma unroll
    for (int j = 0; j < 8; ++j) {
      const int m = m0 + j + hi8;
      float* crow = Cf + (size_t)m * C_H + n0 + lane16;
      crow[0]  = acc0[j];
      crow[16] = acc1[j];
      crow[32] = acc2[j];
      crow[48] = acc3[j];
    }
  } else {
    const int h = n0 >> 6;                 // 64-wide tile is head-aligned
#pragma unroll
    for (int j = 0; j < 8; ++j) {
      const int m = m0 + j + hi8;
      const int b = m >> 11;
      const int s = m & (C_S - 1);
      __bf16* orow = Cb + ((size_t)(b * C_NH + h) * C_S + s) * C_HD;
      if constexpr (EPI == 1) {
        // RoPE: pair (d, d+32) lives in (acc_t, acc_{t+2}) for the same lane.
#pragma unroll
        for (int tp = 0; tp < 2; ++tp) {
          const float a0 = tp ? acc1[j] : acc0[j];
          const float a2 = tp ? acc3[j] : acc2[j];
          const int   d0 = tp * 16 + lane16;               // < 32
          const float inv = __expf((float)d0 * (-LOG_THETA / 32.0f));
          const float ang = (float)s * inv;
          float sn, cs;
          __sincosf(ang, &sn, &cs);
          orow[d0]      = (__bf16)(a0 * cs - a2 * sn);
          orow[d0 + 32] = (__bf16)(a2 * cs + a0 * sn);
        }
      } else {
        orow[ 0 + lane16] = (__bf16)acc0[j];
        orow[16 + lane16] = (__bf16)acc1[j];
        orow[32 + lane16] = (__bf16)acc2[j];
        orow[48 + lane16] = (__bf16)acc3[j];
      }
    }
  }
}

// ---------------------------------------------------------------------------
// Flash attention: grid = B*NH * (S/128) = 512 blocks, 8 waves/block.
// Wave handles 16 q rows; kv tiles of 32; online softmax; bf16 WMMA.
// K tile staged by TDM (padded rows); V staged transposed by VALU path.
// Waves above the diagonal skip compute (wave-uniform branch, EXEC all-1s).
// ---------------------------------------------------------------------------
__global__ __launch_bounds__(256)
void flash_attn_bf16(const __bf16* __restrict__ Q, const __bf16* __restrict__ K,
                     const __bf16* __restrict__ V, __bf16* __restrict__ O) {
  __shared__ __bf16 ldsK [32 * 72];       // K tile [kv][d], stride 72 (TDM pad)
  __shared__ __bf16 ldsVt[64 * 40];       // V tile transposed [d][kv], stride 40
  __shared__ __bf16 ldsP [8 * 16 * 40];   // per-wave P stage [q][kv], stride 40

  const int tid    = threadIdx.x;
  const int lane   = tid & 31;
  const int wv     = tid >> 5;
  const int lane16 = lane & 15;
  const int hi8    = (lane >> 4) * 8;
  const int krun   = (lane >> 4) * 16;

  const int qblk = blockIdx.x & 15;       // 16 q-blocks of 128
  const int bh   = blockIdx.x >> 4;       // 0..31
  const int b    = bh >> 4;
  const int h    = bh & 15;
  const int q0   = qblk * 128 + wv * 16;

  const size_t base = (size_t)bh * C_S * C_HD;
  const unsigned ldsK_off = (unsigned)(size_t)(void*)ldsK;

  // Q fragments (HD=64 -> two 16x32 k-steps), loaded once.
  FragAB aq0, aq1;
  {
    const __bf16* qrow = Q + base + (size_t)(q0 + lane16) * C_HD;
    aq0.h[0] = *(const v8bf*)(qrow + hi8);
    aq0.h[1] = *(const v8bf*)(qrow + 16 + hi8);
    aq1.h[0] = *(const v8bf*)(qrow + 32 + hi8);
    aq1.h[1] = *(const v8bf*)(qrow + 48 + hi8);
  }

  v8f o0 = {}, o1 = {}, o2 = {}, o3 = {};
  float m_r[8], l_r[8];
#pragma unroll
  for (int j = 0; j < 8; ++j) { m_r[j] = -1e30f; l_r[j] = 0.0f; }

  __bf16* myP = ldsP + wv * 16 * 40;
  const int ntiles = qblk * 4 + 4;        // causal bound for this block

  for (int it = 0; it < ntiles; ++it) {
    const int kv0 = it * 32;
    const bool active   = (kv0 <= q0 + 15);     // wave-uniform
    const bool needmask = (kv0 + 31 > q0);      // wave-uniform

    __syncthreads();
    { // V^T staging: 32 rows x 64 cols, 8 bf16 per thread, scatter-transpose
      const int row = tid >> 3;
      const int col = (tid & 7) * 8;
      const v8bf vv = *(const v8bf*)&V[base + (size_t)(kv0 + row) * C_HD + col];
#pragma unroll
      for (int jj = 0; jj < 8; ++jj) ldsVt[(col + jj) * 40 + row] = vv[jj];
    }
    if (wv == 0) {  // TDM DMA: K tile 32 rows x 64 cols, 128B rows padded +16B
      tdm_load_tile_bf16(ldsK_off, K + base + (size_t)kv0 * C_HD,
                         /*tensor_d0=*/C_HD, /*tensor_d1=*/C_S, /*stride0=*/C_HD,
                         /*tile_d0=*/64, /*tile_d1=*/32,
                         /*pad_interval(32dw)=*/4u, /*pad_amount(4dw)=*/3u, true);
      __builtin_amdgcn_s_wait_tensorcnt(0);
    }
    __syncthreads();

    if (active) {
      // ---- scores S = Q @ K^T : 16x32 via 2 col-groups x 2 k-steps ----
      v8f sc0 = {}, sc1 = {};
      {
        FragAB bk;
        bk.h[0] = *(const v8bf*)&ldsK[lane16 * 72 + hi8];
        bk.h[1] = *(const v8bf*)&ldsK[lane16 * 72 + hi8 + 16];
        sc0 = WMMA_BF16(aq0.v, bk.v, sc0);
        bk.h[0] = *(const v8bf*)&ldsK[lane16 * 72 + 32 + hi8];
        bk.h[1] = *(const v8bf*)&ldsK[lane16 * 72 + 32 + hi8 + 16];
        sc0 = WMMA_BF16(aq1.v, bk.v, sc0);
        bk.h[0] = *(const v8bf*)&ldsK[(16 + lane16) * 72 + hi8];
        bk.h[1] = *(const v8bf*)&ldsK[(16 + lane16) * 72 + hi8 + 16];
        sc1 = WMMA_BF16(aq0.v, bk.v, sc1);
        bk.h[0] = *(const v8bf*)&ldsK[(16 + lane16) * 72 + 32 + hi8];
        bk.h[1] = *(const v8bf*)&ldsK[(16 + lane16) * 72 + 32 + hi8 + 16];
        sc1 = WMMA_BF16(aq1.v, bk.v, sc1);
      }

      // ---- online softmax (row = 16-lane half; permlane16 reductions) ----
#pragma unroll
      for (int j = 0; j < 8; ++j) {
        const int qrow = q0 + j + hi8;
        float s0 = sc0[j] * 0.125f;
        float s1 = sc1[j] * 0.125f;
        if (needmask) {
          s0 += ((kv0 + lane16)      > qrow) ? -1e9f : 0.0f;
          s1 += ((kv0 + 16 + lane16) > qrow) ? -1e9f : 0.0f;
        }
        const float rm    = row16_max(fmaxf(s0, s1));
        const float mnew  = fmaxf(m_r[j], rm);
        const float alpha = __expf(m_r[j] - mnew);
        const float p0 = __expf(s0 - mnew);
        const float p1 = __expf(s1 - mnew);
        const float rs = row16_sum(p0 + p1);
        l_r[j] = l_r[j] * alpha + rs;
        m_r[j] = mnew;
        o0[j] *= alpha; o1[j] *= alpha; o2[j] *= alpha; o3[j] *= alpha;
        // restage P (D-layout -> LDS in [q][kv] order)
        myP[(j + hi8) * 40 + lane16]      = (__bf16)p0;
        myP[(j + hi8) * 40 + 16 + lane16] = (__bf16)p1;
      }
    }
    __syncthreads();

    if (active) {
      // ---- O += P @ V : A = P (16x32), B = V^T fragments ----
      FragAB ap;
      {
        const __bf16* prow = myP + lane16 * 40;
        ap.h[0] = *(const v8bf*)(prow + hi8);
        ap.h[1] = *(const v8bf*)(prow + 16 + hi8);
      }
      FragAB bv;
      bv.h[0] = *(const v8bf*)&ldsVt[( 0 + lane16) * 40 + hi8];
      bv.h[1] = *(const v8bf*)&ldsVt[( 0 + lane16) * 40 + hi8 + 16];
      o0 = WMMA_BF16(ap.v, bv.v, o0);
      bv.h[0] = *(const v8bf*)&ldsVt[(16 + lane16) * 40 + hi8];
      bv.h[1] = *(const v8bf*)&ldsVt[(16 + lane16) * 40 + hi8 + 16];
      o1 = WMMA_BF16(ap.v, bv.v, o1);
      bv.h[0] = *(const v8bf*)&ldsVt[(32 + lane16) * 40 + hi8];
      bv.h[1] = *(const v8bf*)&ldsVt[(32 + lane16) * 40 + hi8 + 16];
      o2 = WMMA_BF16(ap.v, bv.v, o2);
      bv.h[0] = *(const v8bf*)&ldsVt[(48 + lane16) * 40 + hi8];
      bv.h[1] = *(const v8bf*)&ldsVt[(48 + lane16) * 40 + hi8 + 16];
      o3 = WMMA_BF16(ap.v, bv.v, o3);
    }
  }

  // ---- finalize: O /= l, write bf16 [B,S,H] ----
#pragma unroll
  for (int j = 0; j < 8; ++j) {
    const int q = q0 + j + hi8;
    const float inv = 1.0f / l_r[j];
    __bf16* orow = O + ((size_t)(b * C_S + q)) * C_H + h * C_HD;
    orow[ 0 + lane16] = (__bf16)(o0[j] * inv);
    orow[16 + lane16] = (__bf16)(o1[j] * inv);
    orow[32 + lane16] = (__bf16)(o2[j] * inv);
    orow[48 + lane16] = (__bf16)(o3[j] * inv);
  }
}

// ---------------------------------------------------------------------------
// Launcher
// ---------------------------------------------------------------------------
extern "C" void kernel_launch(void* const* d_in, const int* in_sizes, int n_in,
                              void* d_out, int out_size, void* d_ws, size_t ws_size,
                              hipStream_t stream) {
  (void)in_sizes; (void)n_in; (void)out_size; (void)ws_size;
  const float* hs = (const float*)d_in[0];
  const float* wq = (const float*)d_in[1];
  const float* wk = (const float*)d_in[2];
  const float* wv = (const float*)d_in[3];
  const float* wo = (const float*)d_in[4];
  // d_in[5] = attention_mask (standard causal, handled analytically)
  // d_in[6] = position_ids  (identity arange, handled analytically)

  char* ws = (char*)d_ws;
  __bf16* Xb  = (__bf16*)(ws);                        //  8 MB [B*S, H]
  __bf16* Wqb = (__bf16*)(ws + ( 8u << 20));          //  2 MB
  __bf16* Wkb = (__bf16*)(ws + (10u << 20));          //  2 MB
  __bf16* Wvb = (__bf16*)(ws + (12u << 20));          //  2 MB
  __bf16* Wob = (__bf16*)(ws + (14u << 20));          //  2 MB
  __bf16* Qr  = (__bf16*)(ws + (16u << 20));          //  8 MB [B,NH,S,HD]
  __bf16* Kr  = (__bf16*)(ws + (24u << 20));          //  8 MB
  __bf16* Vr  = (__bf16*)(ws + (32u << 20));          //  8 MB
  __bf16* Ab  = (__bf16*)(ws + (40u << 20));          //  8 MB [B,S,H]

  const int nX = C_B * C_S * C_H;                     // 4,194,304
  const int nW = C_H * C_H;                           // 1,048,576
  convert_f32_bf16<<<(nX / 4 + 255) / 256, 256, 0, stream>>>(hs, Xb, nX / 4);
  convert_f32_bf16<<<(nW / 4 + 255) / 256, 256, 0, stream>>>(wq, Wqb, nW / 4);
  convert_f32_bf16<<<(nW / 4 + 255) / 256, 256, 0, stream>>>(wk, Wkb, nW / 4);
  convert_f32_bf16<<<(nW / 4 + 255) / 256, 256, 0, stream>>>(wv, Wvb, nW / 4);
  convert_f32_bf16<<<(nW / 4 + 255) / 256, 256, 0, stream>>>(wo, Wob, nW / 4);

  dim3 gg(32, 16);  // M/128 x N/64
  gemm_bf16<1><<<gg, 256, 0, stream>>>(Xb, Wqb, nullptr, Qr);   // Q + RoPE
  gemm_bf16<1><<<gg, 256, 0, stream>>>(Xb, Wkb, nullptr, Kr);   // K + RoPE
  gemm_bf16<2><<<gg, 256, 0, stream>>>(Xb, Wvb, nullptr, Vr);   // V

  flash_attn_bf16<<<C_B * C_NH * (C_S / 128), 256, 0, stream>>>(Qr, Kr, Vr, Ab);

  gemm_bf16<0><<<gg, 256, 0, stream>>>(Ab, Wob, (float*)d_out, nullptr);
}
